// NystromAttention_58884001628801
// MI455X (gfx1250) — matrix operand order repaired and checked by reading, not compiled, not run on hardware
//
#include <hip/hip_runtime.h>

typedef __bf16 bf16_t;
typedef __attribute__((ext_vector_type(16))) __bf16 v16bf;
typedef __attribute__((ext_vector_type(8)))  __bf16 v8bf;
typedef __attribute__((ext_vector_type(2)))  __bf16 v2bf;
typedef __attribute__((ext_vector_type(8)))  float  v8f;

#define APAD 40
#define BPAD 40
#define ATILE (128 * APAD)
#define BTILE (64 * BPAD)

// ---------------- WMMA helpers (layouts per CDNA5 ISA) ----------------

static __device__ __forceinline__ v8f wmma_bf16(v16bf a, v16bf b, v8f c) {
  // 8 args: (neg_a, A, neg_b, B, c_mod, C, reuse_a, reuse_b)
  return __builtin_amdgcn_wmma_f32_16x16x32_bf16(false, a, false, b, (short)0, c, false, false);
}

// A fragment: 16x32 bf16, row-major tile in LDS with row stride APAD.
// lane L<16:  m=L,    k in {0..7, 16..23} ; lane L>=16: m=L-16, k in {8..15, 24..31}
static __device__ __forceinline__ v16bf load_a_frag(const bf16_t* tile, int lane) {
  int m  = lane & 15;
  int kb = (lane & 16) ? 8 : 0;
  const bf16_t* r = tile + m * APAD;
  v8bf lo = *(const v8bf*)(r + kb);
  v8bf hi = *(const v8bf*)(r + 16 + kb);
  v16bf f;
#pragma unroll
  for (int i = 0; i < 8; ++i) { f[i] = lo[i]; f[i + 8] = hi[i]; }
  return f;
}

// B fragment: 32x16 bf16.  LDS tile stored [n][k] with row stride BPAD.
// lane L<16:  n=L, k in {0..7, 16..23} ; lane L>=16: n=L-16, k in {8..15, 24..31}
static __device__ __forceinline__ v16bf load_b_frag(const bf16_t* tile, int lane) {
  int n  = lane & 15;
  int kb = (lane & 16) ? 8 : 0;
  const bf16_t* r = tile + n * BPAD + kb;
  v8bf lo = *(const v8bf*)(r);
  v8bf hi = *(const v8bf*)(r + 16);
  v16bf f;
#pragma unroll
  for (int i = 0; i < 8; ++i) { f[i] = lo[i]; f[i + 8] = hi[i]; }
  return f;
}

// ---------------- Async global->LDS copy (CDNA5 ASYNCcnt path) ----------------

static __device__ __forceinline__ void async_b128(void* lds, const void* gptr) {
  unsigned l = (unsigned)(size_t)lds;   // low 32 bits of generic LDS addr = wave LDS offset
  asm volatile("global_load_async_to_lds_b128 %0, %1, off" :: "v"(l), "v"(gptr) : "memory");
}
static __device__ __forceinline__ void async_wait0() {
  asm volatile("s_wait_asynccnt 0x0" ::: "memory");
}

// ---------------- LDS staging (128 threads, tile M=128 N=64 Kstep=32) ----------------

// f32 -> bf16 converting stage, vectorized (4 x 16B LDS stores per row).
static __device__ __forceinline__ void stage_a_f32(bf16_t* As, const float* A, int lda, int tid) {
  const float* src = A + (size_t)tid * lda;
  v8bf* dst = (v8bf*)(As + tid * APAD);
#pragma unroll
  for (int c = 0; c < 4; ++c) {
    v8bf t;
#pragma unroll
    for (int j = 0; j < 8; ++j) t[j] = (bf16_t)src[c * 8 + j];
    dst[c] = t;
  }
}
// bf16 A tile, pure copy: async DMA, 16B chunks (4 per thread).
static __device__ __forceinline__ void stage_a_bf16_async(bf16_t* As, const bf16_t* A, int lda, int tid) {
#pragma unroll
  for (int i = 0; i < 4; ++i) {
    int c = i * 128 + tid;            // 0..511
    int row = c >> 2, part = c & 3;   // 4 x 8-half chunks per row
    async_b128(As + row * APAD + part * 8, A + (size_t)row * lda + part * 8);
  }
}
// B source [N][K] rows (contiguous k), pure copy: async DMA (2 chunks per thread).
static __device__ __forceinline__ void stage_b_nk_async(bf16_t* Bs, const bf16_t* B, int ldb, int tid) {
#pragma unroll
  for (int i = 0; i < 2; ++i) {
    int c = i * 128 + tid;            // 0..255
    int n = c >> 2, part = c & 3;
    async_b128(Bs + n * BPAD + part * 8, B + (size_t)n * ldb + part * 8);
  }
}
// B source [K][N] row-major f32: transpose into LDS [n][k], pair-packed b32 stores.
static __device__ __forceinline__ void stage_b_kn_f32(bf16_t* Bs, const float* B, int ldb, int tid) {
#pragma unroll
  for (int i = 0; i < 8; ++i) {
    int idx = i * 128 + tid;          // 0..1023 pairs
    int n = idx & 63, kp = idx >> 6;  // kp 0..15
    float lo = B[(size_t)(2 * kp) * ldb + n];
    float hi = B[(size_t)(2 * kp + 1) * ldb + n];
    v2bf t; t[0] = (bf16_t)lo; t[1] = (bf16_t)hi;
    *(v2bf*)(Bs + n * BPAD + 2 * kp) = t;
  }
}
static __device__ __forceinline__ void stage_b_kn_bf16(bf16_t* Bs, const bf16_t* B, int ldb, int tid) {
#pragma unroll
  for (int i = 0; i < 8; ++i) {
    int idx = i * 128 + tid;
    int n = idx & 63, kp = idx >> 6;
    v2bf t;
    t[0] = B[(size_t)(2 * kp) * ldb + n];
    t[1] = B[(size_t)(2 * kp + 1) * ldb + n];
    *(v2bf*)(Bs + n * BPAD + 2 * kp) = t;
  }
}

// One K-step for a wave: wave wid owns rows [wid*32, wid*32+32), all 64 cols.
static __device__ __forceinline__ void mma_step(v8f acc[2][4], const bf16_t* As, const bf16_t* Bs,
                                                int wid, int lane) {
  v16bf a0 = load_a_frag(As + (wid * 32) * APAD, lane);
  v16bf a1 = load_a_frag(As + (wid * 32 + 16) * APAD, lane);
#pragma unroll
  for (int j = 0; j < 4; ++j) {
    v16bf b = load_b_frag(Bs + (j * 16) * BPAD, lane);
    acc[0][j] = wmma_bf16(a0, b, acc[0][j]);
    acc[1][j] = wmma_bf16(a1, b, acc[1][j]);
  }
}

// Double-buffered main loop: issue next tile's loads before current tile's WMMAs.
template <bool ASYNC, typename F>
static __device__ __forceinline__ void gemm_loop(F&& stage, v8f acc[2][4],
                                                 const bf16_t* As, const bf16_t* Bs,
                                                 int steps, int wid, int lane) {
  stage(0, 0);
  if (ASYNC) async_wait0();
  __syncthreads();
  for (int i = 0; i < steps; ++i) {
    if (i + 1 < steps) stage((i + 1) & 1, (i + 1) * 32);
    mma_step(acc, As + (i & 1) * ATILE, Bs + (i & 1) * BTILE, wid, lane);
    if (ASYNC) async_wait0();
    __syncthreads();
  }
}

#define GEMM_PROLOGUE()                                                     \
  __shared__ __align__(16) bf16_t As[2 * ATILE];                            \
  __shared__ __align__(16) bf16_t Bs[2 * BTILE];                            \
  int tid = threadIdx.x, lane = tid & 31, wid = tid >> 5;                   \
  v8f acc[2][4];                                                            \
  {                                                                         \
    v8f zz = {0.f, 0.f, 0.f, 0.f, 0.f, 0.f, 0.f, 0.f};                     \
    for (int i = 0; i < 2; ++i)                                             \
      for (int j = 0; j < 4; ++j) acc[i][j] = zz;                           \
  }

// C-fragment index helpers (32b C/D layout): row = r + 8*(lane>>4), col = lane&15.
#define FRAG_ROW(mi, r) (wid * 32 + (mi) * 16 + (r) + ((lane >> 4) << 3))
#define FRAG_COL(j)     ((j) * 16 + (lane & 15))

// ---------------- Kernel 1: QKV projection + head transpose + q scale ----------------
__global__ __launch_bounds__(128) void k_qkv(const float* __restrict__ x,
                                             const float* __restrict__ w,
                                             bf16_t* __restrict__ qb, bf16_t* __restrict__ kb,
                                             bf16_t* __restrict__ vb) {
  GEMM_PROLOGUE();
  int m0 = blockIdx.x * 128;
  int c0 = blockIdx.y * 64;
  int which = c0 >> 9;              // 0=q 1=k 2=v
  int h = (c0 & 511) >> 6;
  auto stage = [&](int buf, int k0) {
    stage_a_f32(As + buf * ATILE, x + (size_t)m0 * 512 + k0, 512, tid);
    stage_b_kn_f32(Bs + buf * BTILE, w + (size_t)k0 * 1536 + c0, 1536, tid);
  };
  gemm_loop<false>(stage, acc, As, Bs, 16, wid, lane);
  bf16_t* dst = (which == 0) ? qb : (which == 1) ? kb : vb;
  float scale = (which == 0) ? 0.125f : 1.0f;
  for (int mi = 0; mi < 2; ++mi)
    for (int j = 0; j < 4; ++j) {
      v8f a = acc[mi][j];
#pragma unroll
      for (int r = 0; r < 8; ++r) {
        int rowg = m0 + FRAG_ROW(mi, r);
        int b = rowg >> 13, n = rowg & 8191;
        int d = FRAG_COL(j);
        dst[((((size_t)b * 8 + h) * 8192) + n) * 64 + d] = (bf16_t)(a[r] * scale);
      }
    }
}

// ---------------- Kernel 2: landmark means (l=32) ----------------
__global__ void k_landmarks(const bf16_t* __restrict__ q, const bf16_t* __restrict__ k,
                            bf16_t* __restrict__ ql, bf16_t* __restrict__ kl) {
  int bh = blockIdx.x >> 8, mi = blockIdx.x & 255, d = threadIdx.x;
  const bf16_t* qp = q + ((size_t)bh * 8192 + mi * 32) * 64 + d;
  const bf16_t* kp = k + ((size_t)bh * 8192 + mi * 32) * 64 + d;
  float sq = 0.f, sk = 0.f;
  for (int j = 0; j < 32; ++j) { sq += (float)qp[j * 64]; sk += (float)kp[j * 64]; }
  ql[((size_t)bh * 256 + mi) * 64 + d] = (bf16_t)(sq * (1.f / 32.f));
  kl[((size_t)bh * 256 + mi) * 64 + d] = (bf16_t)(sk * (1.f / 32.f));
}

// ---------------- Batched bf16 GEMM, B given as [N][K] rows (async DMA staging) ----------------
__global__ __launch_bounds__(128) void k_gemm_bf16_nk(const bf16_t* __restrict__ A, int lda, size_t sA,
                                                      const bf16_t* __restrict__ B, int ldb, size_t sB,
                                                      float* __restrict__ C, int ldc, size_t sC, int K) {
  GEMM_PROLOGUE();
  int m0 = blockIdx.x * 128, n0 = blockIdx.y * 64, z = blockIdx.z;
  const bf16_t* Ab = A + (size_t)z * sA + (size_t)m0 * lda;
  const bf16_t* Bb = B + (size_t)z * sB + (size_t)n0 * ldb;
  auto stage = [&](int buf, int k0) {
    stage_a_bf16_async(As + buf * ATILE, Ab + k0, lda, tid);
    stage_b_nk_async(Bs + buf * BTILE, Bb + k0, ldb, tid);
  };
  gemm_loop<true>(stage, acc, As, Bs, K >> 5, wid, lane);
  float* Cb = C + (size_t)z * sC;
  for (int mi = 0; mi < 2; ++mi)
    for (int j = 0; j < 4; ++j) {
      v8f a = acc[mi][j];
#pragma unroll
      for (int r = 0; r < 8; ++r)
        Cb[(size_t)(m0 + FRAG_ROW(mi, r)) * ldc + n0 + FRAG_COL(j)] = a[r];
    }
}

// ---------------- Batched bf16 GEMM, B given as [K][N] rows ----------------
__global__ __launch_bounds__(128) void k_gemm_bf16_kn(const bf16_t* __restrict__ A, int lda, size_t sA,
                                                      const bf16_t* __restrict__ B, int ldb, size_t sB,
                                                      float* __restrict__ C, int ldc, size_t sC, int K) {
  GEMM_PROLOGUE();
  int m0 = blockIdx.x * 128, n0 = blockIdx.y * 64, z = blockIdx.z;
  const bf16_t* Ab = A + (size_t)z * sA + (size_t)m0 * lda;
  const bf16_t* Bb = B + (size_t)z * sB + n0;
  auto stage = [&](int buf, int k0) {
    stage_a_bf16_async(As + buf * ATILE, Ab + k0, lda, tid);
    stage_b_kn_bf16(Bs + buf * BTILE, Bb + (size_t)k0 * ldb, ldb, tid);
  };
  gemm_loop<true>(stage, acc, As, Bs, K >> 5, wid, lane);
  float* Cb = C + (size_t)z * sC;
  for (int mi = 0; mi < 2; ++mi)
    for (int j = 0; j < 4; ++j) {
      v8f a = acc[mi][j];
#pragma unroll
      for (int r = 0; r < 8; ++r)
        Cb[(size_t)(m0 + FRAG_ROW(mi, r)) * ldc + n0 + FRAG_COL(j)] = a[r];
    }
}

// ---------------- Batched f32-in GEMM (bf16 WMMA): C = diag*I + alpha*(A@B) ----------------
__global__ __launch_bounds__(128) void k_gemm_f32b(const float* __restrict__ A, int lda, size_t sA,
                                                   const float* __restrict__ B, int ldb, size_t sB,
                                                   float* __restrict__ Cf, bf16_t* __restrict__ Cb16,
                                                   float* __restrict__ Raw, int ldc, size_t sC,
                                                   int K, float diag, float alpha) {
  GEMM_PROLOGUE();
  int m0 = blockIdx.x * 128, n0 = blockIdx.y * 64, z = blockIdx.z;
  const float* Ab = A + (size_t)z * sA + (size_t)m0 * lda;
  const float* Bb = B + (size_t)z * sB + n0;
  auto stage = [&](int buf, int k0) {
    stage_a_f32(As + buf * ATILE, Ab + k0, lda, tid);
    stage_b_kn_f32(Bs + buf * BTILE, Bb + (size_t)k0 * ldb, ldb, tid);
  };
  gemm_loop<false>(stage, acc, As, Bs, K >> 5, wid, lane);
  for (int mi = 0; mi < 2; ++mi)
    for (int j = 0; j < 4; ++j) {
      v8f a = acc[mi][j];
#pragma unroll
      for (int r = 0; r < 8; ++r) {
        int row = m0 + FRAG_ROW(mi, r);
        int col = n0 + FRAG_COL(j);
        size_t idx = (size_t)z * sC + (size_t)row * ldc + col;
        float raw = a[r];
        if (Raw) Raw[idx] = raw;
        float o = alpha * raw + ((row == col) ? diag : 0.f);
        if (Cb16) Cb16[idx] = (bf16_t)o;
        else      Cf[idx] = o;
      }
    }
}

// ---------------- Row softmax: one wave per row ----------------
static __device__ __forceinline__ float wave_max(float v) {
#pragma unroll
  for (int m = 16; m >= 1; m >>= 1) v = fmaxf(v, __shfl_xor(v, m, 32));
  return v;
}
static __device__ __forceinline__ float wave_sum(float v) {
#pragma unroll
  for (int m = 16; m >= 1; m >>= 1) v += __shfl_xor(v, m, 32);
  return v;
}
__global__ __launch_bounds__(256) void k_softmax_rows(const float* in, float* outf,
                                                      bf16_t* outb, int cols) {
  int wid = threadIdx.x >> 5, lane = threadIdx.x & 31;
  size_t row = (size_t)blockIdx.x * 8 + wid;
  const float* p = in + row * (size_t)cols;
  float mx = -3.4e38f;
  for (int c = lane; c < cols; c += 32) mx = fmaxf(mx, p[c]);
  mx = wave_max(mx);
  float s = 0.f;
  for (int c = lane; c < cols; c += 32) s += __expf(p[c] - mx);
  s = wave_sum(s);
  float inv = 1.f / s;
  for (int c = lane; c < cols; c += 32) {
    float v = __expf(p[c] - mx) * inv;
    if (outf) outf[row * (size_t)cols + c] = v;
    if (outb) outb[row * (size_t)cols + c] = (bf16_t)v;
  }
}

// ---------------- pinv init: global max of row/col sums, z0 = X^T / (maxc*maxr) ----------------
__global__ void k_zero(unsigned* cmax) { if (threadIdx.x < 2) cmax[threadIdx.x] = 0u; }

__global__ void k_colsum_max(const float* __restrict__ A2, unsigned* cmax) {
  int bh = blockIdx.x, t = threadIdx.x;
  const float* Mp = A2 + (size_t)bh * 65536;
  float cs = 0.f, rs = 0.f;
  for (int i = 0; i < 256; ++i) cs += Mp[i * 256 + t];
  for (int j = 0; j < 256; ++j) rs += Mp[t * 256 + j];
  atomicMax(&cmax[0], __float_as_uint(cs));
  atomicMax(&cmax[1], __float_as_uint(rs));
}

__global__ void k_zinit(const float* __restrict__ A2, float* __restrict__ Z,
                        const unsigned* __restrict__ cmax) {
  float c = __uint_as_float(cmax[0]) * __uint_as_float(cmax[1]);
  float inv = 1.f / c;
  size_t idx = (size_t)blockIdx.x * 256 + threadIdx.x;
  size_t bh = idx >> 16;
  int rem = (int)(idx & 65535), i = rem >> 8, j = rem & 255;
  Z[idx] = A2[(bh << 16) + (size_t)j * 256 + i] * inv;
}

// ---------------- Depthwise residual conv (kernel 33 along sequence) ----------------
__global__ __launch_bounds__(256) void k_conv(const bf16_t* __restrict__ v,
                                              const float* __restrict__ wconv,
                                              float* __restrict__ oc) {
  __shared__ bf16_t Vs[160 * 64];
  int bh = blockIdx.x, b = bh >> 3, h = bh & 7;
  int i0 = blockIdx.y * 128;
  int tid = threadIdx.x;
  const bf16_t* vp = v + (size_t)bh * 524288;
  for (int i = 0; i < 40; ++i) {
    int idx = i * 256 + tid;
    int r = idx >> 6, d = idx & 63;
    int gi = i0 + r - 16;
    Vs[idx] = (gi >= 0 && gi < 8192) ? vp[(size_t)gi * 64 + d] : (bf16_t)0.f;
  }
  __syncthreads();
  float w[33];
#pragma unroll
  for (int kk = 0; kk < 33; ++kk) w[kk] = wconv[h * 33 + kk];
  int d = tid & 63, rb = tid >> 6;
  for (int jj = 0; jj < 32; ++jj) {
    int r = rb + jj * 4;
    float a = 0.f;
#pragma unroll
    for (int kk = 0; kk < 33; ++kk) a += w[kk] * (float)Vs[(r + kk) * 64 + d];
    oc[((size_t)b * 8192 + i0 + r) * 512 + h * 64 + d] = a;
  }
}

// ---------------- out_core += P1 @ y2  (scatter to [b, n, h*64+d]) ----------------
__global__ __launch_bounds__(128) void k_out_core(const bf16_t* __restrict__ P,
                                                  const bf16_t* __restrict__ y2,
                                                  float* __restrict__ oc) {
  GEMM_PROLOGUE();
  int m0 = blockIdx.x * 128, z = blockIdx.z;
  int b = z >> 3, h = z & 7;
  const bf16_t* Ab = P + (size_t)z * 2097152 + (size_t)m0 * 256;
  const bf16_t* Bb = y2 + (size_t)z * 16384;
  auto stage = [&](int buf, int k0) {
    stage_a_bf16_async(As + buf * ATILE, Ab + k0, 256, tid);
    stage_b_kn_bf16(Bs + buf * BTILE, Bb + (size_t)k0 * 64, 64, tid);
  };
  gemm_loop<true>(stage, acc, As, Bs, 8, wid, lane);
  for (int mi = 0; mi < 2; ++mi)
    for (int j = 0; j < 4; ++j) {
      v8f a = acc[mi][j];
#pragma unroll
      for (int r = 0; r < 8; ++r) {
        int n = m0 + FRAG_ROW(mi, r);
        int d = FRAG_COL(j);
        oc[((size_t)b * 8192 + n) * 512 + h * 64 + d] += a[r];
      }
    }
}

// ---------------- Final: out = out_core @ w_out + b_out ----------------
__global__ __launch_bounds__(128) void k_final(const float* __restrict__ oc,
                                               const float* __restrict__ w,
                                               const float* __restrict__ bias,
                                               float* __restrict__ out) {
  GEMM_PROLOGUE();
  int m0 = blockIdx.x * 128, c0 = blockIdx.y * 64;
  auto stage = [&](int buf, int k0) {
    stage_a_f32(As + buf * ATILE, oc + (size_t)m0 * 512 + k0, 512, tid);
    stage_b_kn_f32(Bs + buf * BTILE, w + (size_t)k0 * 512 + c0, 512, tid);
  };
  gemm_loop<false>(stage, acc, As, Bs, 16, wid, lane);
  for (int mi = 0; mi < 2; ++mi)
    for (int j = 0; j < 4; ++j) {
      v8f a = acc[mi][j];
#pragma unroll
      for (int r = 0; r < 8; ++r) {
        int row = m0 + FRAG_ROW(mi, r);
        int col = c0 + FRAG_COL(j);
        out[(size_t)row * 512 + col] = a[r] + bias[col];
      }
    }
}

// ---------------- Host-side orchestration ----------------
extern "C" void kernel_launch(void* const* d_in, const int* in_sizes, int n_in,
                              void* d_out, int out_size, void* d_ws, size_t ws_size,
                              hipStream_t stream) {
  (void)in_sizes; (void)n_in; (void)out_size; (void)ws_size;
  const float* x      = (const float*)d_in[0];
  const float* w_qkv  = (const float*)d_in[1];
  const float* w_out  = (const float*)d_in[2];
  const float* b_out  = (const float*)d_in[3];
  const float* w_conv = (const float*)d_in[4];
  float* out = (float*)d_out;

  char* pw = (char*)d_ws;
  auto take = [&](size_t bytes) -> char* {
    char* r = pw; pw += (bytes + 255) & ~(size_t)255; return r;
  };
  bf16_t* qb = (bf16_t*)take(33554432);   // [32,8192,64] bf16
  bf16_t* kb = (bf16_t*)take(33554432);
  bf16_t* vb = (bf16_t*)take(33554432);
  bf16_t* ql = (bf16_t*)take(1048576);    // [32,256,64]
  bf16_t* kl = (bf16_t*)take(1048576);
  float*  A2 = (float*)take(8388608);     // softmaxed attn2 (pinv X)
  float*  Z0 = (float*)take(8388608);
  float*  Z1 = (float*)take(8388608);
  float*  T0 = (float*)take(8388608);
  float*  T1 = (float*)take(8388608);
  float*  XZ = (float*)take(8388608);
  float*  y3 = (float*)take(2097152);     // [32,256,64] f32
  bf16_t* y2 = (bf16_t*)take(1048576);    // [32,256,64] bf16
  float*  S  = (float*)take(268435456);   // scores scratch (S3 then S1); later aliased as out_core
  bf16_t* P  = (bf16_t*)take(134217728);  // softmaxed probs bf16 (P3 then P1)
  unsigned* cmax = (unsigned*)take(256);
  float* oc = S;                          // out_core [4,8192,512] aliases dead S region

  // 1) QKV projection (+ transpose to heads, q*scale)
  k_qkv<<<dim3(256, 24), 128, 0, stream>>>(x, w_qkv, qb, kb, vb);
  // 2) landmark means
  k_landmarks<<<8192, 64, 0, stream>>>(qb, kb, ql, kl);
  // 3) attn2 = softmax(ql @ kl^T), kept f32 for pinv
  k_gemm_bf16_nk<<<dim3(2, 4, 32), 128, 0, stream>>>(ql, 64, 16384, kl, 64, 16384,
                                                     A2, 256, 65536, 64);
  k_softmax_rows<<<1024, 256, 0, stream>>>(A2, A2, nullptr, 256);
  // 4) z0 = attn2^T / (max colsum * max rowsum)
  k_zero<<<1, 32, 0, stream>>>(cmax);
  k_colsum_max<<<32, 256, 0, stream>>>(A2, cmax);
  k_zinit<<<8192, 256, 0, stream>>>(A2, Z0, cmax);
  // 5) 6 Newton-Schulz iterations: z = 0.25 z (13I - xz(15I - xz(7I - xz)))
  float* Zc = Z0; float* Zn = Z1;
  for (int it = 0; it < 6; ++it) {
    dim3 g(2, 4, 32);
    k_gemm_f32b<<<g, 128, 0, stream>>>(A2, 256, 65536, Zc, 256, 65536,
                                       T0, nullptr, XZ, 256, 65536, 256, 7.f, -1.f);
    k_gemm_f32b<<<g, 128, 0, stream>>>(XZ, 256, 65536, T0, 256, 65536,
                                       T1, nullptr, nullptr, 256, 65536, 256, 15.f, -1.f);
    k_gemm_f32b<<<g, 128, 0, stream>>>(XZ, 256, 65536, T1, 256, 65536,
                                       T0, nullptr, nullptr, 256, 65536, 256, 13.f, -1.f);
    k_gemm_f32b<<<g, 128, 0, stream>>>(Zc, 256, 65536, T0, 256, 65536,
                                       Zn, nullptr, nullptr, 256, 65536, 256, 0.f, 0.25f);
    float* t = Zc; Zc = Zn; Zn = t;
  }
  // 6) attn3 = softmax(ql @ k^T) -> P3 bf16 ; y3 = P3 @ v
  k_gemm_bf16_nk<<<dim3(2, 128, 32), 128, 0, stream>>>(ql, 64, 16384, kb, 64, 524288,
                                                       S, 8192, 2097152, 64);
  k_softmax_rows<<<1024, 256, 0, stream>>>(S, nullptr, P, 8192);
  k_gemm_bf16_kn<<<dim3(2, 1, 32), 128, 0, stream>>>(P, 8192, 2097152, vb, 64, 524288,
                                                     y3, 64, 16384, 8192);
  // 7) y2 = Z @ y3 (bf16 out)
  k_gemm_f32b<<<dim3(2, 1, 32), 128, 0, stream>>>(Zc, 256, 65536, y3, 64, 16384,
                                                  nullptr, y2, nullptr, 64, 16384, 256, 0.f, 1.f);
  // 8) attn1 = softmax(q @ kl^T) -> P1 bf16
  k_gemm_bf16_nk<<<dim3(64, 4, 32), 128, 0, stream>>>(qb, 64, 524288, kl, 64, 16384,
                                                      S, 256, 2097152, 64);
  k_softmax_rows<<<32768, 256, 0, stream>>>(S, nullptr, P, 256);
  // 9) conv residual writes out_core, then out_core += P1 @ y2
  k_conv<<<dim3(32, 64), 256, 0, stream>>>(vb, w_conv, oc);
  k_out_core<<<dim3(64, 1, 32), 128, 0, stream>>>(P, y2, oc);
  // 10) final projection + bias
  k_final<<<dim3(256, 8), 128, 0, stream>>>(oc, w_out, b_out, out);
}
